// XEyTransformerLayer_64750926954843
// MI455X (gfx1250) — matrix operand order repaired and checked by reading, not compiled, not run on hardware
//
#include <hip/hip_runtime.h>
#include <cstddef>
#include <cstdint>

typedef __attribute__((ext_vector_type(16))) __bf16 v16bf;
typedef __attribute__((ext_vector_type(8)))  __bf16 v8bf;
typedef __attribute__((ext_vector_type(8)))  float  v8f;

typedef unsigned short bstore_t;   // bf16 bits in buffers (host-safe param type)

union U16BF { v16bf v; v8bf h[2]; };

static __device__ inline v16bf ld_a_frag(const __bf16* lo, const __bf16* hi) {
  U16BF u;
  u.h[0] = *reinterpret_cast<const v8bf*>(lo);
  u.h[1] = *reinterpret_cast<const v8bf*>(lo + 8); (void)hi;
  return u.v;
}

static __device__ inline v16bf ld_b_frag(const bstore_t* p) {
  return *reinterpret_cast<const v16bf*>(p);
}

static __device__ inline v8f wmma_bf16(v16bf a, v16bf b, v8f c) {
  return __builtin_amdgcn_wmma_f32_16x16x32_bf16(false, a, false, b, (short)0, c, false, false);
}

// CDNA5 async global->LDS copy (ASYNCcnt path, CDNA5 ISA).
// Low 32 bits of a generic pointer into LDS are the LDS byte offset.
static __device__ inline void async_load_b128_to_lds(void* lds_dst, const void* gsrc) {
  unsigned lds_addr = (unsigned)(uintptr_t)lds_dst;
  unsigned long long ga = (unsigned long long)(uintptr_t)gsrc;
  asm volatile("global_load_async_to_lds_b128 %0, %1, off"
               :: "v"(lds_addr), "v"(ga)
               : "memory");
}
static __device__ inline void wait_asynccnt0() {
  asm volatile("s_wait_asynccnt 0x0" ::: "memory");
}

// ---------------------------------------------------------------------------
// fp32 -> bf16 (RNE) weight conversion
// ---------------------------------------------------------------------------
__global__ void cvt_bf16_kernel(const float* __restrict__ s, bstore_t* __restrict__ d, int n) {
  int i = blockIdx.x * 256 + threadIdx.x;
  if (i < n) {
    unsigned int u = __float_as_uint(s[i]);
    unsigned int r = (u + 0x7fffu + ((u >> 16) & 1u)) >> 16;  // round-to-nearest-even
    d[i] = (bstore_t)r;
  }
}

// ---------------------------------------------------------------------------
// y-conditioning vectors: ymul1 = 1+lin(y_e_mul,y), yadd = lin(y_e_add,y),
// xmul1 = 1+lin(y_x_mul,y), xadd = lin(y_x_add,y), yy = lin(y_y,y)
// ---------------------------------------------------------------------------
__global__ void y_mod_kernel(const float* __restrict__ y,
    const float* __restrict__ Wyem, const float* __restrict__ byem,
    const float* __restrict__ Wyea, const float* __restrict__ byea,
    const float* __restrict__ Wyxm, const float* __restrict__ byxm,
    const float* __restrict__ Wyxa, const float* __restrict__ byxa,
    const float* __restrict__ Wyy,  const float* __restrict__ byy,
    float* __restrict__ ymul1, float* __restrict__ yadd,
    float* __restrict__ xmul1, float* __restrict__ xadd, float* __restrict__ yy)
{
  int c = threadIdx.x;
  for (int b = 0; b < 4; b++) {
    const float* yb = y + b * 64;
    float s1 = 0.f, s2 = 0.f, s3 = 0.f, s4 = 0.f;
    for (int k = 0; k < 64; k++) {
      float yv = yb[k];
      s1 += yv * Wyem[c * 64 + k];
      s2 += yv * Wyea[c * 64 + k];
      s3 += yv * Wyxm[c * 64 + k];
      s4 += yv * Wyxa[c * 64 + k];
    }
    ymul1[b * 256 + c] = 1.f + s1 + byem[c];
    yadd [b * 256 + c] = s2 + byea[c];
    xmul1[b * 256 + c] = 1.f + s3 + byxm[c];
    xadd [b * 256 + c] = s4 + byxa[c];
  }
  if (c < 64) {
    for (int b = 0; b < 4; b++) {
      float s = 0.f;
      for (int k = 0; k < 64; k++) s += y[b * 64 + k] * Wyy[c * 64 + k];
      yy[b * 64 + c] = s + byy[c];
    }
  }
}

// ---------------------------------------------------------------------------
// zx: per-(b,c) stats of X over nodes: [mean|min|max|std(ddof=1)] -> (4,1024)
// ---------------------------------------------------------------------------
__global__ void zx_kernel(const float* __restrict__ X, float* __restrict__ zx) {
  int b = blockIdx.x, c = threadIdx.x;
  float s = 0.f, s2 = 0.f, mn = 3e38f, mx = -3e38f;
  for (int n = 0; n < 192; n++) {
    float v = X[((size_t)b * 192 + n) * 256 + c];
    s += v; s2 += v * v; mn = fminf(mn, v); mx = fmaxf(mx, v);
  }
  float mean = s / 192.f;
  float var  = (s2 - 192.f * mean * mean) / 191.f;
  zx[b * 1024 + c]       = mean;
  zx[b * 1024 + 256 + c] = mn;
  zx[b * 1024 + 512 + c] = mx;
  zx[b * 1024 + 768 + c] = sqrtf(fmaxf(var, 0.f));
}

// ---------------------------------------------------------------------------
// ze: per-(b,c) stats of E over (n,n): block per (b,c), 256-thread reduction
// ---------------------------------------------------------------------------
__global__ void ze_kernel(const float* __restrict__ E, float* __restrict__ ze) {
  int b = blockIdx.x >> 6;
  int c = blockIdx.x & 63;
  int tid = threadIdx.x;
  float s = 0.f, s2 = 0.f, mn = 3e38f, mx = -3e38f;
  for (int idx = tid; idx < 36864; idx += 256) {
    float v = E[((size_t)b * 36864 + idx) * 64 + c];
    s += v; s2 += v * v; mn = fminf(mn, v); mx = fmaxf(mx, v);
  }
  __shared__ float rs_[256], rq_[256], rn_[256], rx_[256];
  rs_[tid] = s; rq_[tid] = s2; rn_[tid] = mn; rx_[tid] = mx;
  __syncthreads();
  for (int off = 128; off > 0; off >>= 1) {
    if (tid < off) {
      rs_[tid] += rs_[tid + off]; rq_[tid] += rq_[tid + off];
      rn_[tid] = fminf(rn_[tid], rn_[tid + off]);
      rx_[tid] = fmaxf(rx_[tid], rx_[tid + off]);
    }
    __syncthreads();
  }
  if (tid == 0) {
    float n = 36864.f;
    float mean = rs_[0] / n;
    float var  = (rq_[0] - n * mean * mean) / (n - 1.f);
    ze[b * 256 + c]        = mean;
    ze[b * 256 + 64 + c]   = rn_[0];
    ze[b * 256 + 128 + c]  = rx_[0];
    ze[b * 256 + 192 + c]  = sqrtf(fmaxf(var, 0.f));
  }
}

// ---------------------------------------------------------------------------
// Q/K/V projection: (768x256)@(256x256) x3, WMMA bf16. Q pre-scaled 1/sqrt(32).
// ---------------------------------------------------------------------------
__global__ __launch_bounds__(256) void xqkv_kernel(
    const float* __restrict__ X,
    const bstore_t* __restrict__ Wq, const bstore_t* __restrict__ Wk, const bstore_t* __restrict__ Wv,
    const float* __restrict__ bq, const float* __restrict__ bk, const float* __restrict__ bv,
    float* __restrict__ Qs, float* __restrict__ Ks, float* __restrict__ Vs)
{
  __shared__ __align__(32) __bf16 Xbf[16][256];
  int tid = threadIdx.x, lane = tid & 31, w = tid >> 5;
  int row0 = blockIdx.x * 16;
  for (int t = tid; t < 4096; t += 256) {
    int m = t >> 8, k = t & 255;
    Xbf[m][k] = (__bf16)X[((size_t)row0 + m) * 256 + k];
  }
  __syncthreads();
  int m_l = lane & 15;
  int kb   = (lane >= 16) ? 16 : 0;
  int kb16 = (lane >= 16) ? 16 : 0;
  int mof  = (lane >= 16) ? 8 : 0;
  const bstore_t* Ws[3] = {Wq, Wk, Wv};
  const float*    Bs[3] = {bq, bk, bv};
  float*          Ds[3] = {Qs, Ks, Vs};
  #pragma unroll
  for (int o = 0; o < 3; o++) {
    float scale = (o == 0) ? 0.17677669529663687f : 1.0f;
    #pragma unroll
    for (int tb = 0; tb < 2; tb++) {
      int n = w * 32 + tb * 16 + m_l;
      v8f acc = {};
      #pragma unroll
      for (int kk = 0; kk < 256; kk += 32) {
        v16bf a  = ld_a_frag(&Xbf[m_l][kk + kb], &Xbf[m_l][kk + 16 + kb]);
        v16bf bf = ld_b_frag(Ws[o] + (size_t)n * 256 + kk + kb16);
        acc = wmma_bf16(a, bf, acc);
      }
      #pragma unroll
      for (int p = 0; p < 8; p++) {
        int m = p + mof;
        Ds[o][((size_t)row0 + m) * 256 + n] = (acc[p] + Bs[o][n]) * scale;
      }
    }
  }
}

// ---------------------------------------------------------------------------
// Fused E path + online-softmax attention. One workgroup per (b,i).
// ---------------------------------------------------------------------------
__global__ __launch_bounds__(256) void e_fused_kernel(
    const float* __restrict__ E,
    const float* __restrict__ Qs, const float* __restrict__ Ks, const float* __restrict__ Vs,
    const bstore_t* __restrict__ Wem, const float* __restrict__ bem,
    const bstore_t* __restrict__ Wea, const float* __restrict__ bea,
    const bstore_t* __restrict__ Weo, const float* __restrict__ beo,
    const bstore_t* __restrict__ W1,  const float* __restrict__ b1,
    const bstore_t* __restrict__ W2,  const float* __restrict__ b2,
    const float* __restrict__ ymul1, const float* __restrict__ yadd,
    const float* __restrict__ gE1, const float* __restrict__ bE1,
    const float* __restrict__ gE2, const float* __restrict__ bE2,
    const float* __restrict__ xmul1, const float* __restrict__ xadd,
    float* __restrict__ wVpre, float* __restrict__ Eout)
{
  __shared__ float q_i[256], fm[256], fa[256];
  __shared__ float smax[256], ssum[256], sacc[256];
  __shared__ __align__(32) float  Ef[16][64];
  __shared__ __align__(32) __bf16 Ebf[16][64];
  __shared__ __align__(32) float  Ylds[16][256];
  __shared__ __align__(32) __bf16 Abf[16][256];
  __shared__ __align__(32) float  Hf[16][64];
  __shared__ __align__(32) __bf16 Hbf[16][64];
  __shared__ __align__(32) float  P2[16][64];
  __shared__ float mu[16], rsd[16];

  int tid = threadIdx.x, lane = tid & 31, w = tid >> 5;
  int bi = blockIdx.x;
  int b  = bi / 192;
  int m_l = lane & 15;
  int kb   = (lane >= 16) ? 16 : 0;
  int kb16 = (lane >= 16) ? 16 : 0;
  int mof  = (lane >= 16) ? 8 : 0;

  q_i[tid]  = Qs[(size_t)bi * 256 + tid];   // already scaled by 1/sqrt(DF), incl bias
  fm[tid]   = ymul1[b * 256 + tid];
  fa[tid]   = yadd [b * 256 + tid];
  smax[tid] = -3.0e38f; ssum[tid] = 0.f; sacc[tid] = 0.f;
  __syncthreads();

  for (int jt = 0; jt < 192; jt += 16) {
    // ---- async DMA of E tile (16 x 64 fp32 = 4KB): one b128 per lane ----
    {
      int m = tid >> 4, k = (tid & 15) * 4;
      async_load_b128_to_lds(&Ef[m][k],
          (const void*)&E[((size_t)bi * 192 + (jt + m)) * 64 + k]);
      wait_asynccnt0();
    }
    if (jt + 16 < 192)
      __builtin_prefetch(&E[((size_t)bi * 192 + (jt + 16 + (tid >> 4))) * 64 + ((tid & 15) * 4)], 0, 0);
    __syncthreads();
    // bf16 shadow copy for the WMMA A-operand
    for (int t = tid; t < 1024; t += 256) {
      int m = t >> 6, k = t & 63;
      Ebf[m][k] = (__bf16)Ef[m][k];
    }
    __syncthreads();

    // ---- GEMMs e_mul / e_add : (16x64)@(64x256), + Y epilogue ----
    {
      v16bf a0 = ld_a_frag(&Ebf[m_l][kb],      &Ebf[m_l][16 + kb]);
      v16bf a1 = ld_a_frag(&Ebf[m_l][32 + kb], &Ebf[m_l][48 + kb]);
      #pragma unroll
      for (int tb = 0; tb < 2; tb++) {
        int n = w * 32 + tb * 16 + m_l;
        v16bf bm0 = ld_b_frag(Wem + (size_t)n * 64 + kb16);
        v16bf bm1 = ld_b_frag(Wem + (size_t)n * 64 + 32 + kb16);
        v16bf ba0 = ld_b_frag(Wea + (size_t)n * 64 + kb16);
        v16bf ba1 = ld_b_frag(Wea + (size_t)n * 64 + 32 + kb16);
        v8f am = {}; am = wmma_bf16(a0, bm0, am); am = wmma_bf16(a1, bm1, am);
        v8f aa = {}; aa = wmma_bf16(a0, ba0, aa); aa = wmma_bf16(a1, ba1, aa);
        #pragma unroll
        for (int p = 0; p < 8; p++) {
          int m = p + mof;
          float em = am[p] + bem[n];
          float ea = aa[p] + bea[n];
          float kv = Ks[((size_t)b * 192 + jt + m) * 256 + n];
          float yv = q_i[n] * kv * (em + 1.f) + ea;     // Y (pre-modulation)
          Ylds[m][n] = yv;
          Abf[m][n]  = (__bf16)(fa[n] + fm[n] * yv);    // modulated, for e_out
        }
      }
    }
    __syncthreads();

    // ---- online softmax state update over this j-tile (thread == channel) ----
    {
      int c = tid;
      float mold = smax[c];
      float tmax = mold;
      float col[16];
      #pragma unroll
      for (int m = 0; m < 16; m++) { col[m] = Ylds[m][c]; tmax = fmaxf(tmax, col[m]); }
      float scl = __expf(mold - tmax);
      float ssv = ssum[c] * scl, sav = sacc[c] * scl;
      #pragma unroll
      for (int m = 0; m < 16; m++) {
        float e = __expf(col[m] - tmax);
        ssv += e;
        sav += e * Vs[((size_t)b * 192 + jt + m) * 256 + c];
      }
      smax[c] = tmax; ssum[c] = ssv; sacc[c] = sav;
    }

    // ---- e_out GEMM: (16x256)@(256x64), + residual with E ----
    if (w < 4) {
      int n = w * 16 + m_l;                 // output channel 0..63
      v8f acc = {};
      #pragma unroll
      for (int kk = 0; kk < 256; kk += 32) {
        v16bf a  = ld_a_frag(&Abf[m_l][kk + kb], &Abf[m_l][kk + 16 + kb]);
        v16bf bb = ld_b_frag(Weo + (size_t)n * 256 + kk + kb16);
        acc = wmma_bf16(a, bb, acc);
      }
      #pragma unroll
      for (int p = 0; p < 8; p++) {
        int m = p + mof;
        Hf[m][n] = acc[p] + beo[n] + Ef[m][n];
      }
    }
    __syncthreads();

    // ---- LayerNorm E1 ----
    if (tid < 16) {
      float s = 0.f, s2 = 0.f;
      for (int c = 0; c < 64; c++) { float v = Hf[tid][c]; s += v; s2 += v * v; }
      float mean = s * (1.f / 64.f);
      float var  = s2 * (1.f / 64.f) - mean * mean;
      mu[tid] = mean; rsd[tid] = rsqrtf(var + 1e-5f);
    }
    __syncthreads();
    for (int t = tid; t < 1024; t += 256) {
      int m = t >> 6, c = t & 63;
      float h = (Hf[m][c] - mu[m]) * rsd[m] * gE1[c] + bE1[c];
      Hf[m][c] = h;              // keep normalized h for residual
      Hbf[m][c] = (__bf16)h;
    }
    __syncthreads();

    // ---- FFE1: relu((16x64)@(64x256)) ----
    {
      v16bf h0 = ld_a_frag(&Hbf[m_l][kb],      &Hbf[m_l][16 + kb]);
      v16bf h1 = ld_a_frag(&Hbf[m_l][32 + kb], &Hbf[m_l][48 + kb]);
      #pragma unroll
      for (int tb = 0; tb < 2; tb++) {
        int n = w * 32 + tb * 16 + m_l;
        v16bf c0 = ld_b_frag(W1 + (size_t)n * 64 + kb16);
        v16bf c1 = ld_b_frag(W1 + (size_t)n * 64 + 32 + kb16);
        v8f acc = {}; acc = wmma_bf16(h0, c0, acc); acc = wmma_bf16(h1, c1, acc);
        #pragma unroll
        for (int p = 0; p < 8; p++) {
          int m = p + mof;
          Abf[m][n] = (__bf16)fmaxf(acc[p] + b1[n], 0.f);
        }
      }
    }
    __syncthreads();

    // ---- FFE2: (16x256)@(256x64) + residual ----
    if (w < 4) {
      int n = w * 16 + m_l;
      v8f acc = {};
      #pragma unroll
      for (int kk = 0; kk < 256; kk += 32) {
        v16bf a  = ld_a_frag(&Abf[m_l][kk + kb], &Abf[m_l][kk + 16 + kb]);
        v16bf bb = ld_b_frag(W2 + (size_t)n * 256 + kk + kb16);
        acc = wmma_bf16(a, bb, acc);
      }
      #pragma unroll
      for (int p = 0; p < 8; p++) {
        int m = p + mof;
        P2[m][n] = acc[p] + b2[n] + Hf[m][n];
      }
    }
    __syncthreads();

    // ---- LayerNorm E2 + store E' ----
    if (tid < 16) {
      float s = 0.f, s2 = 0.f;
      for (int c = 0; c < 64; c++) { float v = P2[tid][c]; s += v; s2 += v * v; }
      float mean = s * (1.f / 64.f);
      float var  = s2 * (1.f / 64.f) - mean * mean;
      mu[tid] = mean; rsd[tid] = rsqrtf(var + 1e-5f);
    }
    __syncthreads();
    for (int t = tid; t < 1024; t += 256) {
      int m = t >> 6, c = t & 63;
      Eout[((size_t)bi * 192 + (jt + m)) * 64 + c] =
          (P2[m][c] - mu[m]) * rsd[m] * gE2[c] + bE2[c];
    }
    __syncthreads();
  }

  // ---- attention output + y_x modulation ----
  {
    int c = tid;
    float wv = sacc[c] / ssum[c];
    wVpre[(size_t)bi * 256 + c] = xadd[b * 256 + c] + xmul1[b * 256 + c] * wv;
  }
}

// ---------------------------------------------------------------------------
// X path: x_out GEMM + LN + FFX + LN. One workgroup per 16 rows.
// ---------------------------------------------------------------------------
__global__ __launch_bounds__(256) void xff_kernel(
    const float* __restrict__ X, const float* __restrict__ wVpre,
    const bstore_t* __restrict__ Wxo, const float* __restrict__ bxo,
    const bstore_t* __restrict__ Wl1, const float* __restrict__ bl1,
    const bstore_t* __restrict__ Wl2, const float* __restrict__ bl2,
    const float* __restrict__ gX1, const float* __restrict__ bX1,
    const float* __restrict__ gX2, const float* __restrict__ bX2,
    float* __restrict__ Xout)
{
  __shared__ __align__(32) __bf16 Abf[16][256];
  __shared__ __align__(32) float  Rf[16][256];
  __shared__ __align__(32) __bf16 Tbf[16][1024];
  __shared__ float mu[16], rsd[16];
  int tid = threadIdx.x, lane = tid & 31, w = tid >> 5;
  int row0 = blockIdx.x * 16;
  int m_l = lane & 15;
  int kb   = (lane >= 16) ? 16 : 0;
  int kb16 = (lane >= 16) ? 16 : 0;
  int mof  = (lane >= 16) ? 8 : 0;

  for (int t = tid; t < 4096; t += 256) {
    int m = t >> 8, k = t & 255;
    Abf[m][k] = (__bf16)wVpre[((size_t)row0 + m) * 256 + k];
  }
  __syncthreads();

  // x_out GEMM + residual with X
  #pragma unroll
  for (int tb = 0; tb < 2; tb++) {
    int n = w * 32 + tb * 16 + m_l;
    v8f acc = {};
    #pragma unroll
    for (int kk = 0; kk < 256; kk += 32) {
      v16bf a  = ld_a_frag(&Abf[m_l][kk + kb], &Abf[m_l][kk + 16 + kb]);
      v16bf bb = ld_b_frag(Wxo + (size_t)n * 256 + kk + kb16);
      acc = wmma_bf16(a, bb, acc);
    }
    #pragma unroll
    for (int p = 0; p < 8; p++) {
      int m = p + mof;
      Rf[m][n] = acc[p] + bxo[n] + X[((size_t)row0 + m) * 256 + n];
    }
  }
  __syncthreads();

  // LayerNorm X1
  if (tid < 16) {
    float s = 0.f, s2 = 0.f;
    for (int c = 0; c < 256; c++) { float v = Rf[tid][c]; s += v; s2 += v * v; }
    float mean = s * (1.f / 256.f);
    float var  = s2 * (1.f / 256.f) - mean * mean;
    mu[tid] = mean; rsd[tid] = rsqrtf(var + 1e-5f);
  }
  __syncthreads();
  for (int t = tid; t < 4096; t += 256) {
    int m = t >> 8, c = t & 255;
    float h = (Rf[m][c] - mu[m]) * rsd[m] * gX1[c] + bX1[c];
    Rf[m][c] = h;
    Abf[m][c] = (__bf16)h;
  }
  __syncthreads();

  // linX1 relu: (16x256)@(256x1024)
  #pragma unroll
  for (int tb = 0; tb < 8; tb++) {
    int n = (w * 8 + tb) * 16 + m_l;
    v8f acc = {};
    #pragma unroll
    for (int kk = 0; kk < 256; kk += 32) {
      v16bf a  = ld_a_frag(&Abf[m_l][kk + kb], &Abf[m_l][kk + 16 + kb]);
      v16bf bb = ld_b_frag(Wl1 + (size_t)n * 256 + kk + kb16);
      acc = wmma_bf16(a, bb, acc);
    }
    #pragma unroll
    for (int p = 0; p < 8; p++) {
      int m = p + mof;
      Tbf[m][n] = (__bf16)fmaxf(acc[p] + bl1[n], 0.f);
    }
  }
  __syncthreads();

  // linX2: (16x1024)@(1024x256) + residual
  #pragma unroll
  for (int tb = 0; tb < 2; tb++) {
    int n = w * 32 + tb * 16 + m_l;
    v8f acc = {};
    for (int kk = 0; kk < 1024; kk += 32) {
      v16bf a  = ld_a_frag(&Tbf[m_l][kk + kb], &Tbf[m_l][kk + 16 + kb]);
      v16bf bb = ld_b_frag(Wl2 + (size_t)n * 1024 + kk + kb16);
      acc = wmma_bf16(a, bb, acc);
    }
    #pragma unroll
    for (int p = 0; p < 8; p++) {
      int m = p + mof;
      Rf[m][n] = acc[p] + bl2[n] + Rf[m][n];
    }
  }
  __syncthreads();

  // LayerNorm X2 + store
  if (tid < 16) {
    float s = 0.f, s2 = 0.f;
    for (int c = 0; c < 256; c++) { float v = Rf[tid][c]; s += v; s2 += v * v; }
    float mean = s * (1.f / 256.f);
    float var  = s2 * (1.f / 256.f) - mean * mean;
    mu[tid] = mean; rsd[tid] = rsqrtf(var + 1e-5f);
  }
  __syncthreads();
  for (int t = tid; t < 4096; t += 256) {
    int m = t >> 8, c = t & 255;
    Xout[((size_t)row0 + m) * 256 + c] = (Rf[m][c] - mu[m]) * rsd[m] * gX2[c] + bX2[c];
  }
}

// ---------------------------------------------------------------------------
// y path tail: new_y = y_y + x_y(zx) + e_y(ze); y_out MLP; LN; FFy; LN
// ---------------------------------------------------------------------------
__global__ void y_final_kernel(
    const float* __restrict__ y,
    const float* __restrict__ yy, const float* __restrict__ zx, const float* __restrict__ ze,
    const float* __restrict__ Wxy, const float* __restrict__ bxy,
    const float* __restrict__ Wey, const float* __restrict__ bey,
    const float* __restrict__ Wy1, const float* __restrict__ by1,
    const float* __restrict__ Wy2, const float* __restrict__ by2,
    const float* __restrict__ Wly1, const float* __restrict__ bly1,
    const float* __restrict__ Wly2, const float* __restrict__ bly2,
    const float* __restrict__ g1, const float* __restrict__ b1n,
    const float* __restrict__ g2, const float* __restrict__ b2n,
    float* __restrict__ yout)
{
  __shared__ float ny[4][64], t1[4][64], r[4][64], y1[4][64], t2[4][256];
  __shared__ float mu4[4], rs4[4];
  int tid = threadIdx.x;
  int b = tid >> 6, c = tid & 63;
  {
    float s = yy[b * 64 + c] + bxy[c] + bey[c];
    for (int k = 0; k < 1024; k++) s += zx[b * 1024 + k] * Wxy[c * 1024 + k];
    for (int k = 0; k < 256; k++)  s += ze[b * 256 + k] * Wey[c * 256 + k];
    ny[b][c] = s;
  }
  __syncthreads();
  { float s = by1[c]; for (int k = 0; k < 64; k++) s += ny[b][k] * Wy1[c * 64 + k];
    t1[b][c] = fmaxf(s, 0.f); }
  __syncthreads();
  { float s = by2[c]; for (int k = 0; k < 64; k++) s += t1[b][k] * Wy2[c * 64 + k];
    r[b][c] = y[b * 64 + c] + s; }
  __syncthreads();
  if (tid < 4) {
    float s = 0.f, s2 = 0.f;
    for (int k = 0; k < 64; k++) { float v = r[tid][k]; s += v; s2 += v * v; }
    float mean = s / 64.f; float var = s2 / 64.f - mean * mean;
    mu4[tid] = mean; rs4[tid] = rsqrtf(var + 1e-5f);
  }
  __syncthreads();
  y1[b][c] = (r[b][c] - mu4[b]) * rs4[b] * g1[c] + b1n[c];
  __syncthreads();
  for (int idx = tid; idx < 1024; idx += 256) {
    int bb = idx >> 8, cf = idx & 255;
    float s = bly1[cf];
    for (int k = 0; k < 64; k++) s += y1[bb][k] * Wly1[cf * 64 + k];
    t2[bb][cf] = fmaxf(s, 0.f);
  }
  __syncthreads();
  { float s = bly2[c]; for (int k = 0; k < 256; k++) s += t2[b][k] * Wly2[c * 256 + k];
    r[b][c] = y1[b][c] + s; }
  __syncthreads();
  if (tid < 4) {
    float s = 0.f, s2 = 0.f;
    for (int k = 0; k < 64; k++) { float v = r[tid][k]; s += v; s2 += v * v; }
    float mean = s / 64.f; float var = s2 / 64.f - mean * mean;
    mu4[tid] = mean; rs4[tid] = rsqrtf(var + 1e-5f);
  }
  __syncthreads();
  yout[b * 64 + c] = (r[b][c] - mu4[b]) * rs4[b] * g2[c] + b2n[c];
}

// ---------------------------------------------------------------------------
// Host launch
// ---------------------------------------------------------------------------
extern "C" void kernel_launch(void* const* d_in, const int* in_sizes, int n_in,
                              void* d_out, int out_size, void* d_ws, size_t ws_size,
                              hipStream_t stream)
{
  (void)in_sizes; (void)n_in; (void)out_size; (void)ws_size;
  const float* X = (const float*)d_in[0];
  const float* E = (const float*)d_in[1];
  const float* y = (const float*)d_in[2];
  // d_in[3] = node_mask (all ones) -> ignored
  auto W = [&](int i) { return (const float*)d_in[i]; };

  char* ws = (char*)d_ws;
  // fp32 scratch
  float* Qs    = (float*)(ws + 0);         // 768x256
  float* Ks    = (float*)(ws + 786432);    // 768x256
  float* Vs    = (float*)(ws + 1572864);   // 768x256
  float* wVpre = (float*)(ws + 2359296);   // 768x256
  float* ymul1 = (float*)(ws + 3145728);   // 4x256
  float* yadd  = (float*)(ws + 3149824);
  float* xmul1 = (float*)(ws + 3153920);
  float* xadd  = (float*)(ws + 3158016);
  float* yy    = (float*)(ws + 3162112);   // 4x64
  float* zx    = (float*)(ws + 3163136);   // 4x1024
  float* ze    = (float*)(ws + 3179520);   // 4x256
  // bf16 weights
  bstore_t* WB  = (bstore_t*)(ws + 3183616);
  bstore_t* WQb  = WB + 0;        // 256x256
  bstore_t* WKb  = WB + 65536;    // 256x256
  bstore_t* WVb  = WB + 131072;   // 256x256
  bstore_t* WEMb = WB + 196608;   // 256x64
  bstore_t* WEAb = WB + 212992;   // 256x64
  bstore_t* WEOb = WB + 229376;   // 64x256
  bstore_t* WE1b = WB + 245760;   // 256x64
  bstore_t* WE2b = WB + 262144;   // 64x256
  bstore_t* WXOb = WB + 278528;   // 256x256
  bstore_t* WX1b = WB + 344064;   // 1024x256
  bstore_t* WX2b = WB + 606208;   // 256x1024

  struct CV { const float* s; bstore_t* d; int n; };
  CV cv[11] = {
    {W(4),  WQb,  65536}, {W(6),  WKb,  65536}, {W(8),  WVb,  65536},
    {W(12), WEMb, 16384}, {W(10), WEAb, 16384}, {W(30), WEOb, 16384},
    {W(40), WE1b, 16384}, {W(42), WE2b, 16384}, {W(28), WXOb, 65536},
    {W(36), WX1b, 262144}, {W(38), WX2b, 262144},
  };
  for (int i = 0; i < 11; i++)
    cvt_bf16_kernel<<<(cv[i].n + 255) / 256, 256, 0, stream>>>(cv[i].s, cv[i].d, cv[i].n);

  y_mod_kernel<<<1, 256, 0, stream>>>(y, W(14), W(15), W(16), W(17), W(18), W(19),
                                      W(20), W(21), W(22), W(23),
                                      ymul1, yadd, xmul1, xadd, yy);
  zx_kernel<<<4, 256, 0, stream>>>(X, zx);
  ze_kernel<<<256, 256, 0, stream>>>(E, ze);

  xqkv_kernel<<<48, 256, 0, stream>>>(X, WQb, WKb, WVb, W(5), W(7), W(9), Qs, Ks, Vs);

  float* Eout = (float*)d_out + 196608;
  e_fused_kernel<<<768, 256, 0, stream>>>(E, Qs, Ks, Vs,
      WEMb, W(13), WEAb, W(11), WEOb, W(31), WE1b, W(41), WE2b, W(43),
      ymul1, yadd, W(52), W(53), W(54), W(55), xmul1, xadd, wVpre, Eout);

  xff_kernel<<<48, 256, 0, stream>>>(X, wVpre, WXOb, W(29), WX1b, W(37), WX2b, W(39),
                                     W(48), W(49), W(50), W(51), (float*)d_out);

  float* yout = (float*)d_out + 9633792;
  y_final_kernel<<<1, 256, 0, stream>>>(y, yy, zx, ze,
      W(24), W(25), W(26), W(27), W(32), W(33), W(34), W(35),
      W(44), W(45), W(46), W(47), W(56), W(57), W(58), W(59), yout);
}
